// AxialAttention_54451595378929
// MI455X (gfx1250) — matrix operand order repaired and checked by reading, not compiled, not run on hardware
//
#include <hip/hip_runtime.h>

// ---------------- types ----------------
typedef __bf16 bf16_t;
typedef __attribute__((ext_vector_type(16))) __bf16 v16bf;
typedef __attribute__((ext_vector_type(8)))  float  v8f;

typedef int v4i __attribute__((vector_size(4 * sizeof(int))));
typedef v4i __attribute__((address_space(1))) g_v4i;   // global (addrspace 1)
typedef v4i __attribute__((address_space(3))) l_v4i;   // LDS    (addrspace 3)

#define NUM_HEADS 12
#define HD   64
#define BB   32
#define HH   28          // grid side
#define NN   785         // 1 + 28*28
#define CC   768
#define C3   2304
#define MTOT (BB * NN)   // 25120

#if defined(__has_builtin)
#  if __has_builtin(__builtin_amdgcn_global_load_async_to_lds_b128)
#    define HAVE_ASYNC_LDS 1
#  endif
#  if __has_builtin(__builtin_amdgcn_s_wait_asynccnt)
#    define HAVE_WAIT_ASYNC_BUILTIN 1
#  endif
#endif

// ---------------- helpers ----------------
__device__ __forceinline__ bf16_t f2bf(float f) {
  union { float f; unsigned u; } v; v.f = f;
  unsigned r = v.u + 0x7fffu + ((v.u >> 16) & 1u);   // RNE
  union { unsigned short s; bf16_t b; } o;
  o.s = (unsigned short)(r >> 16);
  return o.b;
}
__device__ __forceinline__ float bf2f(bf16_t b) {
  union { unsigned short s; bf16_t b; } i; i.b = b;
  union { unsigned u; float f; } o; o.u = ((unsigned)i.s) << 16;
  return o.f;
}
__device__ __forceinline__ v8f v8f_zero() {
  v8f z = {0.f, 0.f, 0.f, 0.f, 0.f, 0.f, 0.f, 0.f};
  return z;
}
__device__ __forceinline__ v8f wmma_bf16(v16bf a, v16bf b, v8f c) {
  return __builtin_amdgcn_wmma_f32_16x16x32_bf16(false, a, false, b, (short)0, c,
                                                 false, false);
}

// 16-byte global -> LDS copy: async on CDNA5 (ASYNCcnt), sync fallback otherwise.
__device__ __forceinline__ void lds_copy16(const bf16_t* g, bf16_t* l) {
#if HAVE_ASYNC_LDS
  __builtin_amdgcn_global_load_async_to_lds_b128((g_v4i*)g, (l_v4i*)l, 0, 0);
#else
  *(uint4*)l = *(const uint4*)g;
#endif
}
__device__ __forceinline__ void wait_async0() {
#if HAVE_WAIT_ASYNC_BUILTIN
  __builtin_amdgcn_s_wait_asynccnt(0);
#elif HAVE_ASYNC_LDS
  asm volatile("s_wait_asynccnt 0" ::: "memory");
#endif
}

// A fragment: 16x32 bf16 tile, row-major, stride ld (elements).
// Lanes 0-15: row = lane, K = [0..15]; lanes 16-31: K = [16..31].
__device__ __forceinline__ v16bf load_frag_a(const bf16_t* base, int ld, int lane) {
  const int m   = lane & 15;
  const int klo = (lane < 16) ? 0 : 16;
  const bf16_t* row = base + (size_t)m * ld;
  v16bf a;
#pragma unroll
  for (int j = 0; j < 8; ++j) a[j]     = row[klo + j];
#pragma unroll
  for (int j = 0; j < 8; ++j) a[8 + j] = row[klo + 8 + j];
  return a;
}

// B fragment: 32x16 bf16 tile stored K-major per column: base[n*ld + k].
// Lane holds column n = lane&15; K half = (lane<16)?0..15:16..31 (contiguous).
__device__ __forceinline__ v16bf load_frag_b(const bf16_t* base, int ld, int lane) {
  const int n     = lane & 15;
  const int khalf = (lane < 16) ? 0 : 16;
  const bf16_t* row = base + (size_t)n * ld + khalf;
  v16bf b;
#pragma unroll
  for (int j = 0; j < 16; ++j) b[j] = row[j];
  return b;
}

// ---------------- prep kernels ----------------
// dst (N x K, n-major) = transpose(src (K x N)) in bf16
__global__ void transpose_convert_f32_bf16(const float* __restrict__ src,
                                           bf16_t* __restrict__ dst,
                                           int K, int N) {
  int idx = blockIdx.x * blockDim.x + threadIdx.x;
  if (idx >= N * K) return;
  int k = idx % K, n = idx / K;
  dst[idx] = f2bf(src[(size_t)k * N + n]);
}

// xseq[b, t] = x[b, perm(t)] where the 28x28 token grid is transposed.
__global__ void permute_convert_x(const float* __restrict__ x,
                                  bf16_t* __restrict__ xseq) {
  size_t idx = (size_t)blockIdx.x * blockDim.x + threadIdx.x;
  const size_t total = (size_t)MTOT * CC;
  if (idx >= total) return;
  int c = (int)(idx % CC);
  size_t rest = idx / CC;
  int t = (int)(rest % NN);
  int b = (int)(rest / NN);
  int src;
  if (t == 0) src = 0;
  else {
    int p = t - 1;
    int i = p / HH, j = p % HH;
    src = 1 + j * HH + i;          // transpose H<->W
  }
  xseq[idx] = f2bf(x[((size_t)b * NN + src) * CC + c]);
}

// ------- tiled bf16 GEMM: C(MxN) = A(MxK) * Bt(NxK)^T, double-buffered LDS -----
// 64x128 tile per 128-thread WG; 4 waves as 2(M)x2(N); per wave 32x64 = 2x4 WMMA.
template <bool HAS_BIAS, bool OUT_BF16>
__global__ __launch_bounds__(128) void gemm_bf16_t(
    const bf16_t* __restrict__ A,  const bf16_t* __restrict__ Bt,
    const float* __restrict__ bias, void* __restrict__ Cptr,
    int M, int N, int K)
{
  __shared__ __align__(16) bf16_t As [2][64][32];    // (m, k)   4 KB / buf
  __shared__ __align__(16) bf16_t Bst[2][128][32];   // (n, k)   8 KB / buf

  const int tid  = threadIdx.x;
  const int lane = tid & 31;
  const int wave = tid >> 5;
  const int m0 = blockIdx.x * 64;
  const int n0 = blockIdx.y * 128;
  const int wm = (wave & 1) * 32;
  const int wn = (wave >> 1) * 64;

  // per-thread copy slots: 16B chunks; row = c>>2, sb = (c&3)*8 elements
  const int rA[2] = { tid >> 2, (tid >> 2) + 32 };           // A rows 0..63
  const int rB[4] = { tid >> 2, (tid >> 2) + 32,
                      (tid >> 2) + 64, (tid >> 2) + 96 };    // B rows 0..127
  const int sb = (tid & 3) * 8;

  const bf16_t* gA[2];
  const bf16_t* gB[4];
#pragma unroll
  for (int i = 0; i < 2; ++i) {
    int gm = m0 + rA[i]; if (gm >= M) gm = M - 1;
    gA[i] = A + (size_t)gm * K + sb;
  }
#pragma unroll
  for (int i = 0; i < 4; ++i)
    gB[i] = Bt + (size_t)(n0 + rB[i]) * K + sb;

  v8f acc[2][4];
#pragma unroll
  for (int i = 0; i < 2; ++i)
#pragma unroll
    for (int j = 0; j < 4; ++j) acc[i][j] = v8f_zero();

  // prime buffer 0
#pragma unroll
  for (int i = 0; i < 2; ++i) lds_copy16(gA[i], &As[0][rA[i]][sb]);
#pragma unroll
  for (int i = 0; i < 4; ++i) lds_copy16(gB[i], &Bst[0][rB[i]][sb]);
  wait_async0();
  __syncthreads();

  int buf = 0;
  for (int k0 = 0; k0 < K; k0 += 32) {
    const bool has_next = (k0 + 32) < K;
    if (has_next) {
      const int kn = k0 + 32;
      const int nb = buf ^ 1;
#pragma unroll
      for (int i = 0; i < 2; ++i) lds_copy16(gA[i] + kn, &As[nb][rA[i]][sb]);
#pragma unroll
      for (int i = 0; i < 4; ++i) lds_copy16(gB[i] + kn, &Bst[nb][rB[i]][sb]);
    }

    v16bf a0 = load_frag_a(&As[buf][wm][0],      32, lane);
    v16bf a1 = load_frag_a(&As[buf][wm + 16][0], 32, lane);
    v16bf bfr[4];
#pragma unroll
    for (int j = 0; j < 4; ++j)
      bfr[j] = load_frag_b(&Bst[buf][wn + j * 16][0], 32, lane);

#pragma unroll
    for (int j = 0; j < 4; ++j) {
      acc[0][j] = wmma_bf16(a0, bfr[j], acc[0][j]);
      acc[1][j] = wmma_bf16(a1, bfr[j], acc[1][j]);
    }

    if (has_next) wait_async0();
    __syncthreads();
    buf ^= 1;
  }

  float*  Cf = (float*)Cptr;
  bf16_t* Cb = (bf16_t*)Cptr;
  const int moff = (lane < 16) ? 0 : 8;
  const int ncol = lane & 15;
#pragma unroll
  for (int ti = 0; ti < 2; ++ti)
#pragma unroll
    for (int tj = 0; tj < 4; ++tj)
#pragma unroll
      for (int v = 0; v < 8; ++v) {
        int gm = m0 + wm + ti * 16 + v + moff;
        int gn = n0 + wn + tj * 16 + ncol;
        if (gm < M) {
          float val = acc[ti][tj][v];
          if (HAS_BIAS) val += bias[gn];
          if (OUT_BF16) Cb[(size_t)gm * N + gn] = f2bf(val);
          else          Cf[(size_t)gm * N + gn] = val;
        }
      }
}

// ---------------- row attention: one wave per (b, row, head) ----------------
__global__ __launch_bounds__(32) void row_attention(
    const bf16_t* __restrict__ qkv,     // (B*785, 2304) xseq order: [q|k|v]
    float* __restrict__ cls_scores,     // (B*12, 785)
    bf16_t* __restrict__ attn_out)      // (B*785, 768) ORIGINAL token order
{
  const int head = blockIdx.x % NUM_HEADS;
  const int r    = (blockIdx.x / NUM_HEADS) % HH;
  const int b    =  blockIdx.x / (NUM_HEADS * HH);
  const int lane = threadIdx.x;

  __shared__ float  S[32][33];
  __shared__ bf16_t P[32][32];
  __shared__ bf16_t Vt[64][32];   // Vt[d][key]

  const size_t base = (size_t)b * NN;
  const float scale = 0.125f;     // 64^-0.5

  // stage V transposed into LDS; keys 29..31 zero
  {
    const int p  = lane;
    const int pc = (p > 28) ? 28 : p;
    const int t  = (pc == 0) ? 0 : (1 + r * HH + (pc - 1));
    const bf16_t* vsrc = qkv + (base + t) * C3 + 2 * CC + head * HD;
    const bool valid = (p < 29);
#pragma unroll 8
    for (int d = 0; d < HD; ++d)
      Vt[d][p] = valid ? vsrc[d] : f2bf(0.0f);
  }

  // S = Q K^T  (29x29 padded to 32x32), hd=64 -> 2 k-steps of 32
  v8f s00 = v8f_zero(), s01 = v8f_zero(), s10 = v8f_zero(), s11 = v8f_zero();
#pragma unroll
  for (int ks = 0; ks < 2; ++ks) {
    v16bf qa[2];
#pragma unroll
    for (int qt = 0; qt < 2; ++qt) {
      int m = (lane & 15) + qt * 16; if (m > 28) m = 28;
      const int t = (m == 0) ? 0 : (1 + r * HH + (m - 1));
      const bf16_t* row = qkv + (base + t) * C3 + head * HD + ks * 32;
      const int klo = (lane < 16) ? 0 : 16;
      v16bf a;
#pragma unroll
      for (int j = 0; j < 8; ++j) a[j]     = row[klo + j];
#pragma unroll
      for (int j = 0; j < 8; ++j) a[8 + j] = row[klo + 8 + j];
      qa[qt] = a;
    }
    v16bf kb[2];
#pragma unroll
    for (int kt = 0; kt < 2; ++kt) {
      int n = (lane & 15) + kt * 16; if (n > 28) n = 28;
      const int t = (n == 0) ? 0 : (1 + r * HH + (n - 1));
      const bf16_t* row = qkv + (base + t) * C3 + CC + head * HD + ks * 32 +
                          ((lane < 16) ? 0 : 16);
      v16bf bb;
#pragma unroll
      for (int j = 0; j < 16; ++j) bb[j] = row[j];
      kb[kt] = bb;
    }
    s00 = wmma_bf16(qa[0], kb[0], s00);
    s01 = wmma_bf16(qa[0], kb[1], s01);
    s10 = wmma_bf16(qa[1], kb[0], s10);
    s11 = wmma_bf16(qa[1], kb[1], s11);
  }

  // spill S to LDS
  {
    const int mo = (lane < 16) ? 0 : 8;
    const int nc = lane & 15;
#pragma unroll
    for (int v = 0; v < 8; ++v) {
      S[v + mo][nc]           = s00[v];
      S[v + mo][16 + nc]      = s01[v];
      S[16 + v + mo][nc]      = s10[v];
      S[16 + v + mo][16 + nc] = s11[v];
    }
  }
  __syncthreads();

  // CLS-query logits (row 0) -> global scratch, xseq-token order
  if (lane >= 1 && lane <= 28)
    cls_scores[(size_t)(b * NUM_HEADS + head) * NN + (1 + r * HH + (lane - 1))] =
        scale * S[0][lane];
  if (lane == 0 && r == 0)
    cls_scores[(size_t)(b * NUM_HEADS + head) * NN + 0] = scale * S[0][0];

  // per-row softmax (rows 1..28 over keys 0..28) -> P (bf16, padded with 0)
  if (lane >= 1 && lane <= 28) {
    float mx = -3.0e38f;
#pragma unroll
    for (int k = 0; k <= 28; ++k) mx = fmaxf(mx, S[lane][k]);
    float e[29];
    float sum = 0.0f;
#pragma unroll
    for (int k = 0; k <= 28; ++k) {
      float t = __expf((S[lane][k] - mx) * scale);
      e[k] = t; sum += t;
    }
    const float inv = 1.0f / sum;
#pragma unroll
    for (int k = 0; k <= 28; ++k) P[lane][k] = f2bf(e[k] * inv);
#pragma unroll
    for (int k = 29; k < 32; ++k) P[lane][k] = f2bf(0.0f);
  } else {
#pragma unroll
    for (int k = 0; k < 32; ++k) P[lane][k] = f2bf(0.0f);
  }
  __syncthreads();

  // O = P x V  (single 32-key k-step; 2 q-tiles x 4 hd-tiles)
  v16bf p0 = load_frag_a(&P[0][0],  32, lane);
  v16bf p1 = load_frag_a(&P[16][0], 32, lane);
  const int nc = lane & 15;
  const int mo = (lane < 16) ? 0 : 8;
#pragma unroll
  for (int ht = 0; ht < 4; ++ht) {
    v16bf vb = load_frag_b(&Vt[ht * 16][0], 32, lane);
    v8f o0 = wmma_bf16(p0, vb, v8f_zero());
    v8f o1 = wmma_bf16(p1, vb, v8f_zero());
    const int dcol = head * HD + ht * 16 + nc;
#pragma unroll
    for (int v = 0; v < 8; ++v) {
      const int q = v + mo;                     // 0..15
      if (q >= 1) {
        const int tokOrig = 1 + (q - 1) * HH + r;   // undo axial transpose
        attn_out[((size_t)b * NN + tokOrig) * CC + dcol] = f2bf(o0[v]);
      }
      const int q2 = q + 16;                    // 16..31
      if (q2 <= 28) {
        const int tokOrig = 1 + (q2 - 1) * HH + r;
        attn_out[((size_t)b * NN + tokOrig) * CC + dcol] = f2bf(o1[v]);
      }
    }
  }
}

// ---------------- global CLS attention: one block per (b, head) --------------
__global__ __launch_bounds__(64) void cls_attention(
    const float* __restrict__ cls_scores,   // (B*12, 785)
    const bf16_t* __restrict__ qkv,         // (B*785, 2304)
    bf16_t* __restrict__ attn_out)          // (B*785, 768) original order
{
  const int bh   = blockIdx.x;
  const int b    = bh / NUM_HEADS;
  const int head = bh % NUM_HEADS;
  const int tid  = threadIdx.x;

  const float* l = cls_scores + (size_t)bh * NN;
  __shared__ float red[64];

  float mx = -3.0e38f;
  for (int k = tid; k < NN; k += 64) mx = fmaxf(mx, l[k]);
  red[tid] = mx; __syncthreads();
  for (int s = 32; s > 0; s >>= 1) {
    if (tid < s) red[tid] = fmaxf(red[tid], red[tid + s]);
    __syncthreads();
  }
  mx = red[0]; __syncthreads();

  float sum = 0.0f;
  for (int k = tid; k < NN; k += 64) sum += __expf(l[k] - mx);
  red[tid] = sum; __syncthreads();
  for (int s = 32; s > 0; s >>= 1) {
    if (tid < s) red[tid] += red[tid + s];
    __syncthreads();
  }
  sum = red[0];
  const float inv = 1.0f / sum;

  // key k in v_cls order == xseq token k
  const int d = tid;   // 0..63
  const bf16_t* vbase = qkv + (size_t)b * NN * C3 + 2 * CC + head * HD + d;
  float acc = 0.0f;
  for (int k = 0; k < NN; ++k)
    acc += __expf(l[k] - mx) * bf2f(vbase[(size_t)k * C3]);

  attn_out[(size_t)b * NN * CC + head * HD + d] = f2bf(acc * inv);  // token 0
}

// ---------------- launch ----------------
extern "C" void kernel_launch(void* const* d_in, const int* in_sizes, int n_in,
                              void* d_out, int out_size, void* d_ws, size_t ws_size,
                              hipStream_t stream) {
  const float* x      = (const float*)d_in[0];
  const float* w_qkv  = (const float*)d_in[1];
  const float* w_proj = (const float*)d_in[2];
  const float* b_proj = (const float*)d_in[3];
  float* out = (float*)d_out;

  char* ws = (char*)d_ws;
  size_t off = 0;
  bf16_t* xseq   = (bf16_t*)(ws + off); off += (size_t)MTOT * CC * 2;       // 38.6 MB
  bf16_t* wqkvT  = (bf16_t*)(ws + off); off += (size_t)CC * C3 * 2;         //  3.5 MB (C3 x CC)
  bf16_t* wprojT = (bf16_t*)(ws + off); off += (size_t)CC * CC * 2;         //  1.2 MB (CC x CC)
  bf16_t* qkv    = (bf16_t*)(ws + off); off += (size_t)MTOT * C3 * 2;       // 116  MB
  float*  cls    = (float*)(ws + off);  off += (size_t)BB * NUM_HEADS * NN * 4;
  bf16_t* attn   = (bf16_t*)(ws + off); off += (size_t)MTOT * CC * 2;       // 38.6 MB

  // 1. prep: permuted x -> bf16, weights -> transposed bf16 (n-major)
  {
    size_t n = (size_t)MTOT * CC;
    permute_convert_x<<<dim3((unsigned)((n + 255) / 256)), 256, 0, stream>>>(x, xseq);
  }
  {
    int n = CC * C3;
    transpose_convert_f32_bf16<<<dim3((n + 255) / 256), 256, 0, stream>>>(
        w_qkv, wqkvT, CC, C3);
  }
  {
    int n = CC * CC;
    transpose_convert_f32_bf16<<<dim3((n + 255) / 256), 256, 0, stream>>>(
        w_proj, wprojT, CC, CC);
  }

  // 2. QKV GEMM: (25120 x 768) @ (768 x 2304) -> bf16
  {
    dim3 grid((MTOT + 63) / 64, C3 / 128);
    gemm_bf16_t<false, true><<<grid, 128, 0, stream>>>(xseq, wqkvT, nullptr,
                                                       (void*)qkv, MTOT, C3, CC);
  }

  // 3. row attention (+ CLS logit spill)
  row_attention<<<dim3(BB * HH * NUM_HEADS), 32, 0, stream>>>(qkv, cls, attn);

  // 4. global CLS attention
  cls_attention<<<dim3(BB * NUM_HEADS), 64, 0, stream>>>(cls, qkv, attn);

  // 5. output projection with bias: (25120 x 768) @ (768 x 768) -> fp32 d_out
  {
    dim3 grid((MTOT + 63) / 64, CC / 128);
    gemm_bf16_t<true, false><<<grid, 128, 0, stream>>>(attn, wprojT, b_proj,
                                                       (void*)out, MTOT, CC, CC);
  }
}
